// Attention_3788161155562
// MI455X (gfx1250) — matrix operand order repaired and checked by reading, not compiled, not run on hardware
//
#include <hip/hip_runtime.h>
#include <hip/hip_bf16.h>
#include <float.h>
#include <math.h>
#include <stdint.h>

#define B_    4
#define N_    1024
#define DIM_  1024
#define H_    16
#define DH_   64
#define NMEM_ 4
#define TOPK_ 64
#define J_    1068
#define JP_   1088      // padded to multiple of 64 for WMMA tiles / K loop
#define JROW_ 1092      // LDS row stride in floats: 1092 % 64 == 4 -> conflict-free head gathers
#define QK_SCALE_ 10.0f

typedef _Float16 f16;
typedef __attribute__((ext_vector_type(8)))  _Float16 hv8;
typedef __attribute__((ext_vector_type(16))) _Float16 hv16;
typedef __attribute__((ext_vector_type(8)))  float    fv8;

// ---------------------------------------------------------------------------
// NT GEMM:  C[m,n] = scale * sum_k A[m,k] * Bt[n,k]
// f16 in, f32 accumulate via v_wmma_f32_16x16x32_f16.
// Block = 128 threads (4 waves, 2x2); each wave owns a 32x32 C tile
// (2x2 register blocking: 4 fragment loads -> 4 WMMAs).
// grid = (N/64, M/64, batch). M,N multiples of 64; K multiple of 32.
// ---------------------------------------------------------------------------
__device__ __forceinline__ hv16 load_a_frag(const f16* __restrict__ rowp, int kk, int lane) {
    const int kg = (lane >> 4) * 8;                 // lanes 16..31 start at K+8
    hv8 lo = *reinterpret_cast<const hv8*>(rowp + kk + kg);
    hv8 hi = *reinterpret_cast<const hv8*>(rowp + kk + kg + 16);
    hv16 r;
#pragma unroll
    for (int t = 0; t < 8; ++t) { r[t] = lo[t]; r[t + 8] = hi[t]; }
    return r;
}

__device__ __forceinline__ hv16 load_b_frag(const f16* __restrict__ rowp, int kk, int lane) {
    // B 32x16: same K order as A (lanes 16..31 start at K+8, second half at +16)
    const int kg = (lane >> 4) * 8;
    hv8 lo = *reinterpret_cast<const hv8*>(rowp + kk + kg);
    hv8 hi = *reinterpret_cast<const hv8*>(rowp + kk + kg + 16);
    hv16 r;
#pragma unroll
    for (int t = 0; t < 8; ++t) { r[t] = lo[t]; r[t + 8] = hi[t]; }
    return r;
}

__device__ __forceinline__ void store_tile(float* __restrict__ Cp, long ldc,
                                           int tileM, int tileN, int lane,
                                           const fv8& acc, float scale) {
    const int col  = tileN + (lane & 15);
    const int rowb = tileM + 8 * (lane >> 4);       // D layout: VGPR r -> M = r (+8 hi lanes)
#pragma unroll
    for (int r = 0; r < 8; ++r)
        Cp[(long)(rowb + r) * ldc + col] = acc[r] * scale;
}

__global__ __launch_bounds__(128) void k_gemm_nt(
    const f16* __restrict__ A, long lda, long sA,
    const f16* __restrict__ Bt, long ldb, long sB,
    float* __restrict__ C, long ldc, long sC,
    int M, int N, int K, float scale)
{
    const int lane  = threadIdx.x & 31;
    const int wave  = threadIdx.x >> 5;
    const int tileM = blockIdx.y * 64 + (wave >> 1) * 32;   // wave-uniform
    const int tileN = blockIdx.x * 64 + (wave & 1) * 32;
    if (tileN >= N || tileM >= M) return;                   // uniform: EXEC stays all-1s
    const long z = blockIdx.z;

    const f16* Arow0 = A  + z * sA + (long)(tileM + (lane & 15)) * lda;
    const f16* Arow1 = Arow0 + 16 * lda;
    const f16* Brow0 = Bt + z * sB + (long)(tileN + (lane & 15)) * ldb;
    const f16* Brow1 = Brow0 + 16 * ldb;

    fv8 acc00 = {}, acc01 = {}, acc10 = {}, acc11 = {};
    for (int kk = 0; kk < K; kk += 32) {
        hv16 a0 = load_a_frag(Arow0, kk, lane);
        hv16 a1 = load_a_frag(Arow1, kk, lane);
        hv16 b0 = load_b_frag(Brow0, kk, lane);
        hv16 b1 = load_b_frag(Brow1, kk, lane);
        acc00 = __builtin_amdgcn_wmma_f32_16x16x32_f16(false, a0, false, b0, (short)0, acc00, false, false);
        acc01 = __builtin_amdgcn_wmma_f32_16x16x32_f16(false, a0, false, b1, (short)0, acc01, false, false);
        acc10 = __builtin_amdgcn_wmma_f32_16x16x32_f16(false, a1, false, b0, (short)0, acc10, false, false);
        acc11 = __builtin_amdgcn_wmma_f32_16x16x32_f16(false, a1, false, b1, (short)0, acc11, false, false);
    }

    float* Cp = C + z * sC;
    store_tile(Cp, ldc, tileM,      tileN,      lane, acc00, scale);
    store_tile(Cp, ldc, tileM,      tileN + 16, lane, acc01, scale);
    store_tile(Cp, ldc, tileM + 16, tileN,      lane, acc10, scale);
    store_tile(Cp, ldc, tileM + 16, tileN + 16, lane, acc11, scale);
}

// ---------------------------------------------------------------------------
// Elementwise helpers
// ---------------------------------------------------------------------------
__global__ void k_f32_to_f16(const float* __restrict__ s, f16* __restrict__ d, long n) {
    long i = (long)blockIdx.x * blockDim.x + threadIdx.x;
    if (i < n) d[i] = (f16)s[i];
}

// dst[c*K + k] = (f16) src[k*C + c]   (transpose + downconvert)
__global__ void k_transpose_f16(const float* __restrict__ s, f16* __restrict__ d, int K, int C) {
    long i = (long)blockIdx.x * blockDim.x + threadIdx.x;
    if (i >= (long)K * C) return;
    int c = (int)(i % C);
    int k = (int)(i / C);
    d[(long)c * K + k] = (f16)s[i];
}

__device__ __forceinline__ float block_sum64(float v, volatile float* red) {
    int t = threadIdx.x;
    red[t] = v; __syncthreads();
#pragma unroll
    for (int s = 32; s > 0; s >>= 1) {
        if (t < s) red[t] = red[t] + red[t + s];
        __syncthreads();
    }
    float r = red[0];
    __syncthreads();
    return r;
}

// l2-normalize q,k (with mem_k prepended, zero-padded to JP_), build vt = v^T.
// grid (JP_, H, B), block 64 (one thread per d).
__global__ void k_repack(const float* __restrict__ qkvg,
                         const float* __restrict__ mem_k, const float* __restrict__ mem_v,
                         f16* __restrict__ qh, f16* __restrict__ kh, f16* __restrict__ vt)
{
    const int j = blockIdx.x, h = blockIdx.y, b = blockIdx.z, d = threadIdx.x;
    __shared__ float red[64];

    // ---- q (rows i = j < N_) ----
    float qv = (j < N_) ? qkvg[(long)(b * N_ + j) * (4 * DIM_) + h * DH_ + d] : 0.f;
    float ss = block_sum64(qv * qv, red);
    if (j < N_) {
        float rn = 1.0f / fmaxf(sqrtf(ss), 1e-12f);
        qh[((long)(b * H_ + h) * N_ + j) * DH_ + d] = (f16)(qv * rn);
    }

    // ---- k (mem slots, projected keys, zero padding) ----
    float kv = 0.f;
    if (j < NMEM_)       kv = mem_k[((long)h * NMEM_ + j) * DH_ + d];
    else if (j < J_)     kv = qkvg[(long)(b * N_ + j - NMEM_) * (4 * DIM_) + DIM_ + h * DH_ + d];
    ss = block_sum64(kv * kv, red);
    float rn = 1.0f / fmaxf(sqrtf(ss), 1e-12f);
    kh[((long)(b * H_ + h) * JP_ + j) * DH_ + d] = (f16)(kv * rn);

    // ---- v transposed: vt[b,h,d,j] ----
    float vv = 0.f;
    if (j < NMEM_)       vv = mem_v[((long)h * NMEM_ + j) * DH_ + d];
    else if (j < J_)     vv = qkvg[(long)(b * N_ + j - NMEM_) * (4 * DIM_) + 2 * DIM_ + h * DH_ + d];
    vt[((long)(b * H_ + h) * DH_ + d) * JP_ + j] = (f16)vv;
}

// ---------------------------------------------------------------------------
// Fused: pre talking-heads mix + shifted-causal mask + exact top-64 threshold
// + softmax + post talking-heads mix, all in LDS for one (b, i) query row.
// Reads dots once, writes attnh once (saves ~1.1 GB of HBM traffic vs the
// unfused 3-kernel version). One block = 256 threads (8 waves) per (b,i);
// 16 head-rows of 1088 f32 staged in LDS (~70 KB of the WGP's 320 KB).
// Global->LDS staging uses the CDNA5 async-LDS path (ASYNCcnt).
// ---------------------------------------------------------------------------
__global__ __launch_bounds__(256) void k_mix_topk_softmax(
    const float* __restrict__ dots, const float* __restrict__ Wpre,
    const float* __restrict__ Wpost, f16* __restrict__ attnh)
{
    __shared__ float rows[H_][JROW_];
    __shared__ float wpre[H_ * H_];
    __shared__ float wpost[H_ * H_];
    const int t = threadIdx.x;
    const int i = blockIdx.x;                 // query position
    const int b = blockIdx.y;
    const long hs = (long)N_ * JP_;
    const float* gbase = dots + (long)b * H_ * hs + (long)i * JP_;

    wpre[t]  = Wpre[t];                       // blockDim == 256 == H_*H_
    wpost[t] = Wpost[t];

    // --- async global->LDS copy of the 16 head rows for this (b,i) ---
    const int CH = (JP_ * 4) / 16;            // 272 x 16B chunks per row
    unsigned lbase = (unsigned)(uintptr_t)(&rows[0][0]);
    for (int idx = t; idx < H_ * CH; idx += 256) {
        int g = idx / CH, c = idx % CH;
        const float* ga = gbase + (long)g * hs + c * 4;
        unsigned loff = lbase + (unsigned)(g * (JROW_ * 4) + c * 16);
        asm volatile("global_load_async_to_lds_b128 %0, %1, off"
                     :: "v"(loff), "v"(ga) : "memory");
    }
    asm volatile("s_wait_asynccnt 0x0" ::: "memory");
    __syncthreads();

    // --- pre talking-heads mix + mask, in place, column-parallel ---
    for (int j = t; j < JP_; j += 256) {
        float g[H_];
#pragma unroll
        for (int gg = 0; gg < H_; ++gg) g[gg] = rows[gg][j];
        const bool masked = (j > i + NMEM_);  // shifted causal (covers padded cols)
#pragma unroll
        for (int hh = 0; hh < H_; ++hh) {
            float s = -FLT_MAX;
            if (!masked) {
                s = 0.f;
#pragma unroll
                for (int gg = 0; gg < H_; ++gg) s += wpre[hh * H_ + gg] * g[gg];
            }
            rows[hh][j] = s;
        }
    }
    __syncthreads();

    // --- per-head exact top-64 (radix select on monotonic uint map) + softmax ---
    const int wv = t >> 5, lane = t & 31;
    for (int h = wv; h < H_; h += 8) {        // 8 waves, 2 heads each, wave-local
        float* r = rows[h];
        unsigned T = 0u;
        for (int bit = 31; bit >= 0; --bit) {
            unsigned cand = T | (1u << bit);
            int cnt = 0;
            for (int j = lane; j < JP_; j += 32) {
                unsigned u = __float_as_uint(r[j]);
                u = (u & 0x80000000u) ? ~u : (u | 0x80000000u);
                cnt += (u >= cand) ? 1 : 0;
            }
#pragma unroll
            for (int s = 16; s > 0; s >>= 1) cnt += __shfl_xor(cnt, s, 32);
            if (cnt >= TOPK_) T = cand;
        }
        float vk = __uint_as_float((T & 0x80000000u) ? (T & 0x7FFFFFFFu) : ~T);

        float m = -FLT_MAX;
        for (int j = lane; j < JP_; j += 32) m = fmaxf(m, r[j]);
#pragma unroll
        for (int s = 16; s > 0; s >>= 1) m = fmaxf(m, __shfl_xor(m, s, 32));

        float sum = 0.f;
        for (int j = lane; j < JP_; j += 32) {
            float f = r[j];
            float p = (f < vk) ? 0.f : __expf(f - m);
            r[j] = p; sum += p;
        }
#pragma unroll
        for (int s = 16; s > 0; s >>= 1) sum += __shfl_xor(sum, s, 32);
        float inv = 1.0f / sum;
        for (int j = lane; j < JP_; j += 32) r[j] *= inv;
    }
    __syncthreads();

    // --- post talking-heads mix -> f16 A-operand for attn@V ---
    f16* obase = attnh + (long)b * H_ * hs + (long)i * JP_;
    for (int j = t; j < JP_; j += 256) {
        float g[H_];
#pragma unroll
        for (int gg = 0; gg < H_; ++gg) g[gg] = rows[gg][j];
#pragma unroll
        for (int hh = 0; hh < H_; ++hh) {
            float s = 0.f;
#pragma unroll
            for (int gg = 0; gg < H_; ++gg) s += wpost[hh * H_ + gg] * g[gg];
            obase[(long)hh * hs + j] = (f16)s;
        }
    }
}

// head_scale * sigmoid(gates) gating + [b,h,i,d] -> [b*N+i, h*DH+d] repack to f16
__global__ void k_gate(const float* __restrict__ outp, const float* __restrict__ qkvg,
                       const float* __restrict__ bg, const float* __restrict__ hsc,
                       f16* __restrict__ og)
{
    long idx = (long)blockIdx.x * blockDim.x + threadIdx.x;
    if (idx >= (long)B_ * N_ * (H_ * DH_)) return;
    int  c = (int)(idx % (H_ * DH_));
    long r = idx / (H_ * DH_);
    int h = c / DH_, d = c % DH_;
    int b = (int)(r / N_), i = (int)(r % N_);
    float o  = outp[((long)(b * H_ + h) * N_ + i) * DH_ + d];
    float z  = qkvg[r * (4 * DIM_) + 3 * DIM_ + c] + bg[c];
    float sg = 1.0f / (1.0f + __expf(-z));
    og[r * (H_ * DH_) + c] = (f16)(o * hsc[h] * sg);
}

// ---------------------------------------------------------------------------
extern "C" void kernel_launch(void* const* d_in, const int* in_sizes, int n_in,
                              void* d_out, int out_size, void* d_ws, size_t ws_size,
                              hipStream_t stream)
{
    (void)in_sizes; (void)n_in; (void)out_size; (void)ws_size;
    const float* x    = (const float*)d_in[0];
    const float* Wq   = (const float*)d_in[1];
    const float* Wk   = (const float*)d_in[2];
    const float* Wv   = (const float*)d_in[3];
    const float* Wpre = (const float*)d_in[4];
    const float* Wpost= (const float*)d_in[5];
    const float* memk = (const float*)d_in[6];
    const float* memv = (const float*)d_in[7];
    const float* hsc  = (const float*)d_in[8];
    const float* Wg   = (const float*)d_in[9];
    const float* bg   = (const float*)d_in[10];
    const float* Wo   = (const float*)d_in[11];
    float* out = (float*)d_out;

    char* base = (char*)d_ws; size_t off = 0;
    auto alloc = [&](size_t bytes) -> void* {
        void* p = base + off;
        off = (off + bytes + 255) & ~(size_t)255;
        return p;
    };
    const long R = (long)B_ * N_;                     // 4096 rows
    f16*   xh    = (f16*)  alloc((size_t)R * DIM_ * 2);
    f16*   wt    = (f16*)  alloc((size_t)4 * DIM_ * DIM_ * 2);   // [c=4096][k=1024]
    f16*   wot   = (f16*)  alloc((size_t)DIM_ * DIM_ * 2);       // Wo^T
    float* qkvg  = (float*)alloc((size_t)R * 4 * DIM_ * 4);
    f16*   qh    = (f16*)  alloc((size_t)B_ * H_ * N_  * DH_ * 2);
    f16*   kh    = (f16*)  alloc((size_t)B_ * H_ * JP_ * DH_ * 2);
    f16*   vt    = (f16*)  alloc((size_t)B_ * H_ * DH_ * JP_ * 2);
    float* dots  = (float*)alloc((size_t)B_ * H_ * N_ * JP_ * 4);
    f16*   attnh = (f16*)  alloc((size_t)B_ * H_ * N_ * JP_ * 2);
    float* outp  = (float*)alloc((size_t)B_ * H_ * N_ * DH_ * 4);
    f16*   og    = (f16*)  alloc((size_t)R * (H_ * DH_) * 2);

    // 1) f16 conversions / weight transposes
    k_f32_to_f16<<<(R * DIM_ + 255) / 256, 256, 0, stream>>>(x, xh, R * DIM_);
    const long WN = (long)DIM_ * DIM_;
    k_transpose_f16<<<(WN + 255) / 256, 256, 0, stream>>>(Wq, wt + 0 * WN, DIM_, DIM_);
    k_transpose_f16<<<(WN + 255) / 256, 256, 0, stream>>>(Wk, wt + 1 * WN, DIM_, DIM_);
    k_transpose_f16<<<(WN + 255) / 256, 256, 0, stream>>>(Wv, wt + 2 * WN, DIM_, DIM_);
    k_transpose_f16<<<(WN + 255) / 256, 256, 0, stream>>>(Wg, wt + 3 * WN, DIM_, DIM_);
    k_transpose_f16<<<(WN + 255) / 256, 256, 0, stream>>>(Wo, wot, DIM_, DIM_);

    // 2) fused Q|K|V|G projection GEMM: [4096,1024] x [1024,4096]
    k_gemm_nt<<<dim3(4 * DIM_ / 64, R / 64, 1), 128, 0, stream>>>(
        xh, DIM_, 0, wt, DIM_, 0, qkvg, 4 * DIM_, 0, (int)R, 4 * DIM_, DIM_, 1.0f);

    // 3) l2norm + mem concat + v transpose
    k_repack<<<dim3(JP_, H_, B_), 64, 0, stream>>>(qkvg, memk, memv, qh, kh, vt);

    // 4) QK^T per (b,h): [1024,64] x [64,1088], scaled by 10
    k_gemm_nt<<<dim3(JP_ / 64, N_ / 64, B_ * H_), 128, 0, stream>>>(
        qh, DH_, (long)N_ * DH_, kh, DH_, (long)JP_ * DH_,
        dots, JP_, (long)N_ * JP_, N_, JP_, DH_, QK_SCALE_);

    // 5) fused pre-mix + mask + top-64 + softmax + post-mix -> f16
    k_mix_topk_softmax<<<dim3(N_, B_), 256, 0, stream>>>(dots, Wpre, Wpost, attnh);

    // 6) attn @ V per (b,h): [1024,1088] x [1088,64]
    k_gemm_nt<<<dim3(DH_ / 64, N_ / 64, B_ * H_), 128, 0, stream>>>(
        attnh, JP_, (long)N_ * JP_, vt, JP_, (long)DH_ * JP_,
        outp, DH_, (long)N_ * DH_, N_, DH_, JP_, 1.0f);

    // 7) head scale * sigmoid gate, repack to [4096, 1024] f16
    k_gate<<<(R * H_ * DH_ + 255) / 256, 256, 0, stream>>>(outp, qkvg, bg, hsc, og);

    // 8) output projection: [4096,1024] x [1024,1024] -> d_out (f32)
    k_gemm_nt<<<dim3(DIM_ / 64, R / 64, 1), 128, 0, stream>>>(
        og, H_ * DH_, 0, wot, DIM_, 0, out, DIM_, 0, (int)R, DIM_, H_ * DH_, 1.0f);
}
